// MatchingBlock_54709293416653
// MI455X (gfx1250) — matrix operand order repaired and checked by reading, not compiled, not run on hardware
//
#include <hip/hip_runtime.h>
#include <hip/hip_bf16.h>

// ---------------------------------------------------------------------------
// CDNA5 (gfx1250) fused dual-modality cross-attention, bf16 WMMA pipeline
// with TDM (tensor_load_to_lds) staging and software-pipelined WMMA loops.
// ---------------------------------------------------------------------------

typedef __attribute__((ext_vector_type(16))) __bf16 v16bf;
typedef __attribute__((ext_vector_type(8)))  __bf16 v8bf;
typedef __attribute__((ext_vector_type(8)))  float  v8f;
typedef __attribute__((ext_vector_type(4)))  unsigned int u32x4;
typedef __attribute__((ext_vector_type(8)))  int i32x8;
typedef __attribute__((ext_vector_type(4)))  int i32x4;

#define SCALE_F 0.044194173824159216f  // 1/sqrt(512)
#define NEG_F   (-1e9f)

__device__ __forceinline__ v8f wmma_bf16(v16bf a, v16bf b, v8f c) {
    return __builtin_amdgcn_wmma_f32_16x16x32_bf16(
        false, a, false, b, (short)0, c, false, false);
}

// A-matrix (16x32 bf16), row-major bf16 tile, leading-dim ldk.
// lane L -> row (L&15); elems 0..15 = K kk0+((L>=16)?16:0)+i.
__device__ __forceinline__ v16bf load_A(const __bf16* base, int ldk, int kk0, int lane) {
    int row = lane & 15;
    int k0  = kk0 + ((lane >> 4) << 4);
    const __bf16* p = base + row * ldk + k0;
    v8bf lo = *(const v8bf*)p;
    v8bf hi = *(const v8bf*)(p + 8);
    return __builtin_shufflevector(lo, hi, 0,1,2,3,4,5,6,7,8,9,10,11,12,13,14,15);
}

// B-matrix (32x16 bf16) from K-major buffer bt[col*ldk+k].
// lanes 0-15 -> col lane, K kk0..+15; lanes 16-31 -> K +16. One 32B load.
__device__ __forceinline__ v16bf load_B(const __bf16* bt, int ldk, int col0, int kk0, int lane) {
    int col = col0 + (lane & 15);
    int k0  = kk0 + ((lane >> 4) << 4);
    return *(const v16bf*)(bt + col * ldk + k0);
}

// ---------------------------------------------------------------------------
// TDM: DMA a 2D fp32 tile (rows x rowElems, row stride strideElems) into LDS.
// Builds D# group0/group1 per CDNA5 ISA 8.3/8.4; issued by one wave.
// ---------------------------------------------------------------------------
__device__ __forceinline__ void tdm_load_2d_f32(const float* gptr, void* lptr,
                                                unsigned rowElems, unsigned rows,
                                                unsigned strideElems) {
    unsigned lds = (unsigned)(unsigned long long)lptr;       // LDS byte offset
    unsigned long long ga = (unsigned long long)gptr;
    u32x4 g0;
    g0[0] = 1u;                                              // count=1, user mode
    g0[1] = lds;                                             // lds_addr [63:32]
    g0[2] = (unsigned)(ga & 0xFFFFFFFFu);                    // global_addr lo
    g0[3] = (unsigned)((ga >> 32) & 0x1FFFFFFu) | (2u << 30);// addr hi | type=2
    i32x8 g1;
    g1[0] = (int)(2u << 16);                                 // data_size=4B
    g1[1] = (int)(rowElems << 16);                           // tensor_dim0 lo16
    g1[2] = (int)((rowElems >> 16) | (rows << 16));          // dim0 hi | dim1 lo
    g1[3] = (int)((rows >> 16) | (rowElems << 16));          // dim1 hi | tile_dim0
    g1[4] = (int)rows;                                       // tile_dim1, tile_dim2=0
    g1[5] = (int)strideElems;                                // tensor_dim0_stride lo32
    g1[6] = 0;
    g1[7] = 0;
    i32x4 z4 = {0, 0, 0, 0};
#if __clang_major__ >= 23
    i32x8 z8 = {0, 0, 0, 0, 0, 0, 0, 0};
    __builtin_amdgcn_tensor_load_to_lds(g0, g1, z4, z4, z8, 0);
#else
    __builtin_amdgcn_tensor_load_to_lds(g0, g1, z4, z4, 0);
#endif
}

// ---------------------------------------------------------------------------
// K0: transpose+convert one 512x512 fp32 weight (K,N) -> bf16 (N,K)
// ---------------------------------------------------------------------------
__global__ __launch_bounds__(256) void wt_kernel(const float* __restrict__ W,
                                                 __bf16* __restrict__ WT) {
    int e = blockIdx.x * 256 + threadIdx.x;
    int k = e >> 9, n = e & 511;
    WT[n * 512 + k] = (__bf16)W[k * 512 + n];
}

// ---------------------------------------------------------------------------
// K1: K = X@Wk -> (B,N,HD) bf16 ; V = X@Wv -> (B,F,N) bf16 (transposed)
// grid (N/16, B), block 128.  Two pipelined k-passes (K then V).
// ---------------------------------------------------------------------------
__global__ __launch_bounds__(128) void proj_kv_kernel(
        const float*  __restrict__ X,    // (N,B,A)
        const __bf16* __restrict__ WkT,  // (HD,A) bf16
        const __bf16* __restrict__ WvT,  // (F,A)  bf16
        __bf16* __restrict__ Kbf,        // (B,N,HD)
        __bf16* __restrict__ Vt) {       // (B,F,N)
    __shared__ float  sStage[16 * 512];
    __shared__ __bf16 sX[16 * 512];
    int ntile = blockIdx.x, b = blockIdx.y;
    int tid = threadIdx.x, lane = tid & 31, wave = tid >> 5;

    if (tid == 0) {
        tdm_load_2d_f32(X + ((ntile * 16) * 8 + b) * 512, sStage, 512, 16, 8 * 512);
        __builtin_amdgcn_s_wait_tensorcnt((short)0);
    }
    __syncthreads();
    for (int e = tid; e < 16 * 512; e += 128) sX[e] = (__bf16)sStage[e];
    __syncthreads();

    int rowoff = (lane >> 4) << 3, cl = lane & 15;
    int colb = wave * 8 * 16;

    {   // pass 1: K projection, ping-pong pipelined over K=512
        v8f acc[8] = {};
        v16bf b0[8], b1[8];
#pragma unroll
        for (int c = 0; c < 8; ++c) b0[c] = load_B(WkT, 512, colb + c * 16, 0, lane);
        for (int kk = 0; kk < 512; kk += 64) {
            v16bf a0 = load_A(sX, 512, kk, lane);
#pragma unroll
            for (int c = 0; c < 8; ++c) b1[c] = load_B(WkT, 512, colb + c * 16, kk + 32, lane);
#pragma unroll
            for (int c = 0; c < 8; ++c) acc[c] = wmma_bf16(a0, b0[c], acc[c]);
            v16bf a1 = load_A(sX, 512, kk + 32, lane);
#pragma unroll
            for (int c = 0; c < 8; ++c) b0[c] = load_B(WkT, 512, colb + c * 16, kk + 64, lane);
#pragma unroll
            for (int c = 0; c < 8; ++c) acc[c] = wmma_bf16(a1, b1[c], acc[c]);
        }
#pragma unroll
        for (int c = 0; c < 8; ++c) {
            int col = colb + c * 16 + cl;
#pragma unroll
            for (int r = 0; r < 8; ++r)
                Kbf[(b * 128 + ntile * 16 + r + rowoff) * 512 + col] = (__bf16)acc[c][r];
        }
    }
    {   // pass 2: V projection, store transposed (F,N)
        v8f acc[8] = {};
        v16bf b0[8], b1[8];
#pragma unroll
        for (int c = 0; c < 8; ++c) b0[c] = load_B(WvT, 512, colb + c * 16, 0, lane);
        for (int kk = 0; kk < 512; kk += 64) {
            v16bf a0 = load_A(sX, 512, kk, lane);
#pragma unroll
            for (int c = 0; c < 8; ++c) b1[c] = load_B(WvT, 512, colb + c * 16, kk + 32, lane);
#pragma unroll
            for (int c = 0; c < 8; ++c) acc[c] = wmma_bf16(a0, b0[c], acc[c]);
            v16bf a1 = load_A(sX, 512, kk + 32, lane);
#pragma unroll
            for (int c = 0; c < 8; ++c) b0[c] = load_B(WvT, 512, colb + c * 16, kk + 64, lane);
#pragma unroll
            for (int c = 0; c < 8; ++c) acc[c] = wmma_bf16(a1, b1[c], acc[c]);
        }
#pragma unroll
        for (int c = 0; c < 8; ++c) {
            int col = colb + c * 16 + cl;
#pragma unroll
            for (int r = 0; r < 8; ++r)
                Vt[(b * 512 + col) * 128 + ntile * 16 + r + rowoff] = (__bf16)acc[c][r];
        }
    }
}

// ---------------------------------------------------------------------------
// K2: q = Y@Wq (LDS bf16), logit = q@K^T * SCALE -> fp32 ws (B,T,N)
// grid (T/16, B), block 128.
// ---------------------------------------------------------------------------
__global__ __launch_bounds__(128) void q_logits_kernel(
        const float*  __restrict__ Y,    // (T,B,F)
        const __bf16* __restrict__ WqT,  // (HD,F)
        const __bf16* __restrict__ Kbf,  // (B,N,HD)
        float* __restrict__ logit) {     // (B,T,N)
    __shared__ float  sStage[16 * 512];
    __shared__ __bf16 sY[16 * 512];
    __shared__ __bf16 sQ[16 * 512];
    int ttile = blockIdx.x, b = blockIdx.y;
    int tid = threadIdx.x, lane = tid & 31, wave = tid >> 5;
    int t0 = ttile * 16;

    if (tid == 0) {
        tdm_load_2d_f32(Y + (t0 * 8 + b) * 512, sStage, 512, 16, 8 * 512);
        __builtin_amdgcn_s_wait_tensorcnt((short)0);
    }
    __syncthreads();
    for (int e = tid; e < 16 * 512; e += 128) sY[e] = (__bf16)sStage[e];
    __syncthreads();

    int rowoff = (lane >> 4) << 3, cl = lane & 15;
    int colb = wave * 8 * 16;

    {   // q projection: 16x512, K=512, pipelined
        v8f acc[8] = {};
        v16bf b0[8], b1[8];
#pragma unroll
        for (int c = 0; c < 8; ++c) b0[c] = load_B(WqT, 512, colb + c * 16, 0, lane);
        for (int kk = 0; kk < 512; kk += 64) {
            v16bf a0 = load_A(sY, 512, kk, lane);
#pragma unroll
            for (int c = 0; c < 8; ++c) b1[c] = load_B(WqT, 512, colb + c * 16, kk + 32, lane);
#pragma unroll
            for (int c = 0; c < 8; ++c) acc[c] = wmma_bf16(a0, b0[c], acc[c]);
            v16bf a1 = load_A(sY, 512, kk + 32, lane);
#pragma unroll
            for (int c = 0; c < 8; ++c) b0[c] = load_B(WqT, 512, colb + c * 16, kk + 64, lane);
#pragma unroll
            for (int c = 0; c < 8; ++c) acc[c] = wmma_bf16(a1, b1[c], acc[c]);
        }
#pragma unroll
        for (int c = 0; c < 8; ++c) {
            int col = colb + c * 16 + cl;
#pragma unroll
            for (int r = 0; r < 8; ++r)
                sQ[(r + rowoff) * 512 + col] = (__bf16)acc[c][r];
        }
    }
    __syncthreads();

    {   // logits: 16x128, K=512; B operand = rows of Kbf (K-major already)
        const __bf16* Kb = Kbf + b * 128 * 512;
        int n0 = wave * 2 * 16;
        v8f acc[2] = {};
        v16bf b0[2], b1[2];
        b0[0] = load_B(Kb, 512, n0, 0, lane);
        b0[1] = load_B(Kb, 512, n0 + 16, 0, lane);
        for (int kk = 0; kk < 512; kk += 64) {
            v16bf a0 = load_A(sQ, 512, kk, lane);
            b1[0] = load_B(Kb, 512, n0, kk + 32, lane);
            b1[1] = load_B(Kb, 512, n0 + 16, kk + 32, lane);
            acc[0] = wmma_bf16(a0, b0[0], acc[0]);
            acc[1] = wmma_bf16(a0, b0[1], acc[1]);
            v16bf a1 = load_A(sQ, 512, kk + 32, lane);
            b0[0] = load_B(Kb, 512, n0, kk + 64, lane);
            b0[1] = load_B(Kb, 512, n0 + 16, kk + 64, lane);
            acc[0] = wmma_bf16(a1, b1[0], acc[0]);
            acc[1] = wmma_bf16(a1, b1[1], acc[1]);
        }
#pragma unroll
        for (int c = 0; c < 2; ++c) {
            int n = n0 + c * 16 + cl;
#pragma unroll
            for (int r = 0; r < 8; ++r)
                logit[(b * 4096 + t0 + r + rowoff) * 128 + n] = acc[c][r] * SCALE_F;
        }
    }
}

// ---------------------------------------------------------------------------
// K3: fused cross-modal bias + sim + pad-mask + softmax + PV + residual
// grid (T/16, B), block 128.
// ---------------------------------------------------------------------------
__global__ __launch_bounds__(128) void attn_out_kernel(
        const float*  __restrict__ logit_self,   // (B,T,N)
        const float*  __restrict__ logit_other,  // (B,T,N)
        const float*  __restrict__ mask,         // (B,N,1)
        const float*  __restrict__ align_w,      // scalar
        const float*  __restrict__ Y,            // (T,B,F) residual
        const __bf16* __restrict__ Vt,           // (B,F,N)
        float* __restrict__ out,                 // (T,B,F)
        float* __restrict__ sim) {               // (B,N,T)
    __shared__ float  sL[16 * 128];
    __shared__ __bf16 sA[16 * 128];
    __shared__ float  redmax[16 * 8], redsum[16 * 8];
    __shared__ float  rowmax[16], rowinv[16];
    int ttile = blockIdx.x, b = blockIdx.y;
    int tid = threadIdx.x, lane = tid & 31, wave = tid >> 5;
    int t0 = ttile * 16;
    float sig = 1.f / (1.f + __expf(-align_w[0]));

    {   // phase 1: thread owns key column n = tid
        int n = tid;
        bool pad = (mask[b * 128 + n] == 0.f);
        float* sp = sim + ((size_t)b * 128 + n) * 4096 + t0;
#pragma unroll
        for (int r = 0; r < 16; ++r) {
            int idx = (b * 4096 + t0 + r) * 128 + n;
            float lnm = logit_self[idx] + sig * logit_other[idx];
            sp[r] = lnm;
            sL[r * 128 + n] = pad ? NEG_F : lnm;
        }
    }
    __syncthreads();
    {   // phase 2: row max
        int r = tid >> 3, sub = tid & 7;
        float m = -3.0e38f;
#pragma unroll
        for (int i = 0; i < 16; ++i) m = fmaxf(m, sL[r * 128 + sub * 16 + i]);
        redmax[r * 8 + sub] = m;
    }
    __syncthreads();
    if (tid < 16) {
        float m = redmax[tid * 8];
#pragma unroll
        for (int i = 1; i < 8; ++i) m = fmaxf(m, redmax[tid * 8 + i]);
        rowmax[tid] = m;
    }
    __syncthreads();
    {   // phase 3: row sum of exp
        int r = tid >> 3, sub = tid & 7;
        float m = rowmax[r], s = 0.f;
#pragma unroll
        for (int i = 0; i < 16; ++i) s += __expf(sL[r * 128 + sub * 16 + i] - m);
        redsum[r * 8 + sub] = s;
    }
    __syncthreads();
    if (tid < 16) {
        float s = 0.f;
#pragma unroll
        for (int i = 0; i < 8; ++i) s += redsum[tid * 8 + i];
        rowinv[tid] = 1.f / s;
    }
    __syncthreads();
    {   // phase 4: normalized attn -> bf16 LDS
        int r = tid >> 3, sub = tid & 7;
        float m = rowmax[r], inv = rowinv[r];
#pragma unroll
        for (int i = 0; i < 16; ++i) {
            int n = sub * 16 + i;
            sA[r * 128 + n] = (__bf16)(__expf(sL[r * 128 + n] - m) * inv);
        }
    }
    __syncthreads();

    // phase 5: out = attn(16x128) @ V(128x512) + Y ; K=128, pipelined
    int rowoff = (lane >> 4) << 3, cl = lane & 15;
    int colb = wave * 8 * 16;
    const __bf16* Vb = Vt + (size_t)b * 512 * 128;
    v8f acc[8] = {};
    v16bf b0[8], b1[8];
#pragma unroll
    for (int c = 0; c < 8; ++c) b0[c] = load_B(Vb, 128, colb + c * 16, 0, lane);
#pragma unroll
    for (int kk = 0; kk < 128; kk += 64) {
        v16bf a0 = load_A(sA, 128, kk, lane);
#pragma unroll
        for (int c = 0; c < 8; ++c) b1[c] = load_B(Vb, 128, colb + c * 16, kk + 32, lane);
#pragma unroll
        for (int c = 0; c < 8; ++c) acc[c] = wmma_bf16(a0, b0[c], acc[c]);
        v16bf a1 = load_A(sA, 128, kk + 32, lane);
#pragma unroll
        for (int c = 0; c < 8; ++c) b0[c] = load_B(Vb, 128, colb + c * 16, kk + 64, lane);
#pragma unroll
        for (int c = 0; c < 8; ++c) acc[c] = wmma_bf16(a1, b1[c], acc[c]);
    }
#pragma unroll
    for (int c = 0; c < 8; ++c) {
        int f = colb + c * 16 + cl;
#pragma unroll
        for (int r = 0; r < 8; ++r) {
            int t = t0 + r + rowoff;
            size_t o = ((size_t)t * 8 + b) * 512 + f;
            out[o] = acc[c][r] + Y[o];
        }
    }
}

// ---------------------------------------------------------------------------
// K4: task head — 40 dots of length 512 (+bias)
// ---------------------------------------------------------------------------
__global__ __launch_bounds__(64) void task_kernel(const float* __restrict__ node,
                                                  const float* __restrict__ W,
                                                  const float* __restrict__ bias,
                                                  float* __restrict__ outp) {
    int i = threadIdx.x;
    if (i >= 40) return;
    const float* p = node + i * 512;
    float s = 0.f;
    for (int a = 0; a < 512; ++a) s += p[a] * W[a];
    outp[i] = s + bias[0];
}

// ---------------------------------------------------------------------------
extern "C" void kernel_launch(void* const* d_in, const int* in_sizes, int n_in,
                              void* d_out, int out_size, void* d_ws, size_t ws_size,
                              hipStream_t stream) {
    (void)in_sizes; (void)n_in; (void)out_size; (void)ws_size;
    const float* v_task_node = (const float*)d_in[0];
    const float* v_proto     = (const float*)d_in[1];
    const float* v_frame     = (const float*)d_in[2];
    const float* t_task_node = (const float*)d_in[3];
    const float* t_proto     = (const float*)d_in[4];
    const float* t_frame     = (const float*)d_in[5];
    const float* v_mask      = (const float*)d_in[6];
    const float* t_mask      = (const float*)d_in[7];
    const float* v_Wq        = (const float*)d_in[8];
    const float* v_Wk        = (const float*)d_in[9];
    const float* v_Wv        = (const float*)d_in[10];
    const float* v_align     = (const float*)d_in[11];
    const float* v_task_W    = (const float*)d_in[12];
    const float* v_task_b    = (const float*)d_in[13];
    const float* t_Wq        = (const float*)d_in[14];
    const float* t_Wk        = (const float*)d_in[15];
    const float* t_Wv        = (const float*)d_in[16];
    const float* t_align     = (const float*)d_in[17];
    const float* t_task_W    = (const float*)d_in[18];
    const float* t_task_b    = (const float*)d_in[19];

    char* ws = (char*)d_ws;
    __bf16* WT    = (__bf16*)ws;                 // 6 x 512x512 bf16   (3 MB)
    __bf16* Kbf   = (__bf16*)(ws + 3145728);     // 2 x (B,N,HD) bf16  (2 MB)
    __bf16* Vt    = (__bf16*)(ws + 5242880);     // 2 x (B,F,N) bf16   (2 MB)
    float*  logit = (float*) (ws + 7340032);     // 2 x (B,T,N) f32    (32 MB)

    const size_t WSZ  = 512 * 512;
    const size_t KVSZ = 8 * 128 * 512;
    const size_t LGSZ = (size_t)8 * 4096 * 128;

    wt_kernel<<<1024, 256, 0, stream>>>(v_Wq, WT + 0 * WSZ);
    wt_kernel<<<1024, 256, 0, stream>>>(v_Wk, WT + 1 * WSZ);
    wt_kernel<<<1024, 256, 0, stream>>>(v_Wv, WT + 2 * WSZ);
    wt_kernel<<<1024, 256, 0, stream>>>(t_Wq, WT + 3 * WSZ);
    wt_kernel<<<1024, 256, 0, stream>>>(t_Wk, WT + 4 * WSZ);
    wt_kernel<<<1024, 256, 0, stream>>>(t_Wv, WT + 5 * WSZ);

    proj_kv_kernel<<<dim3(8, 8), 128, 0, stream>>>(v_proto, WT + 1 * WSZ, WT + 2 * WSZ,
                                                   Kbf, Vt);
    proj_kv_kernel<<<dim3(8, 8), 128, 0, stream>>>(t_proto, WT + 4 * WSZ, WT + 5 * WSZ,
                                                   Kbf + KVSZ, Vt + KVSZ);

    q_logits_kernel<<<dim3(256, 8), 128, 0, stream>>>(v_frame, WT + 0 * WSZ, Kbf, logit);
    q_logits_kernel<<<dim3(256, 8), 128, 0, stream>>>(t_frame, WT + 3 * WSZ, Kbf + KVSZ,
                                                      logit + LGSZ);

    float* out   = (float*)d_out;
    float* v_out = out;
    float* v_ta  = out + 16777216;
    float* v_sim = out + 16777256;
    float* t_out = out + 20971560;
    float* t_ta  = out + 37748776;
    float* t_sim = out + 37748816;

    attn_out_kernel<<<dim3(256, 8), 128, 0, stream>>>(logit, logit + LGSZ, v_mask,
                                                      v_align, v_frame, Vt, v_out, v_sim);
    attn_out_kernel<<<dim3(256, 8), 128, 0, stream>>>(logit + LGSZ, logit, t_mask,
                                                      t_align, t_frame, Vt + KVSZ, t_out, t_sim);

    task_kernel<<<1, 64, 0, stream>>>(v_task_node, v_task_W, v_task_b, v_ta);
    task_kernel<<<1, 64, 0, stream>>>(t_task_node, t_task_W, t_task_b, t_ta);
}
